// VisualEmbeddingLayer_13907104104696
// MI455X (gfx1250) — matrix operand order repaired and checked by reading, not compiled, not run on hardware
//
#include <hip/hip_runtime.h>
#include <hip/hip_bf16.h>
#include <hip/hip_fp16.h>
#include <float.h>

typedef __attribute__((ext_vector_type(16))) _Float16 v16h;
typedef __attribute__((ext_vector_type(8)))  _Float16 v8h;
typedef __attribute__((ext_vector_type(8)))  float    v8f;
typedef __attribute__((ext_vector_type(4)))  int      v4i;

// Problem constants (match reference setup_inputs)
constexpr int B    = 512;
constexpr int N    = 193;
constexpr int D    = 512;
constexpr int E    = 2048;
constexpr int G4   = 4;              // group size
constexpr int KTOK = 96;             // (N-1)/2
constexpr int H    = 1024;           // MLP hidden
constexpr int GRP  = B / G4;         // 128 groups
constexpr int RG   = G4 * KTOK;      // 384 rows per group

// ---------------------------------------------------------------------------
// CDNA5 async global->LDS staging (ASYNCcnt path), with portable fallback
// ---------------------------------------------------------------------------
#if defined(__gfx1250__) && __has_builtin(__builtin_amdgcn_global_load_async_to_lds_b128)
#define USE_ASYNC_LDS 1
#else
#define USE_ASYNC_LDS 0
#endif

__device__ __forceinline__ void async_cp16(void* lds_dst, const void* gsrc)
{
#if USE_ASYNC_LDS
    __builtin_amdgcn_global_load_async_to_lds_b128((v4i*)gsrc, (v4i*)lds_dst, 0, 0);
#else
    *(v8h*)lds_dst = *(const v8h*)gsrc;
#endif
}

__device__ __forceinline__ void stage_wait()
{
#if USE_ASYNC_LDS
#if __has_builtin(__builtin_amdgcn_s_wait_asynccnt)
    __builtin_amdgcn_s_wait_asynccnt(0);
#else
    asm volatile("s_wait_asynccnt 0x0" ::: "memory");
#endif
#endif
    __syncthreads();
}

// ---------------------------------------------------------------------------
// K1: top-k via rank (stable descending, ties -> lower index first)
// ---------------------------------------------------------------------------
__global__ void topk_kernel(const float* __restrict__ atten, int* __restrict__ idx)
{
    __shared__ float s[N];
    const int b = blockIdx.x;
    const float* arow = atten + (size_t)b * N * N;   // atten[b, 0, :]
    for (int j = threadIdx.x; j < N; j += blockDim.x)
        s[j] = (j == 0) ? -1.0f : arow[j];
    __syncthreads();
    for (int i = threadIdx.x; i < N; i += blockDim.x) {
        const float si = s[i];
        int rank = 0;
        for (int j = 0; j < N; ++j) {
            const float sj = s[j];
            rank += ((sj > si) || (sj == si && j < i)) ? 1 : 0;
        }
        if (rank < KTOK) idx[b * KTOK + rank] = i;
    }
}

// ---------------------------------------------------------------------------
// K2: gather selected tokens + row l2norm, emit f16 activations X (R x D)
// ---------------------------------------------------------------------------
__global__ void gather_norm_kernel(const float* __restrict__ bf,
                                   const int* __restrict__ idx,
                                   _Float16* __restrict__ Xh)
{
    const int row = blockIdx.x;           // 0 .. B*KTOK-1
    const int b = row / KTOK, t = row % KTOK;
    const int tok = idx[b * KTOK + t];
    const float* src = bf + ((size_t)b * N + tok) * D;

    __shared__ float red[4];
    const int tid = threadIdx.x;          // 128 threads
    float v[4];
    float ss = 0.f;
    for (int e = 0; e < 4; ++e) {
        v[e] = src[tid + e * 128];
        ss += v[e] * v[e];
    }
    for (int o = 16; o > 0; o >>= 1) ss += __shfl_xor(ss, o, 32);
    if ((tid & 31) == 0) red[tid >> 5] = ss;
    __syncthreads();
    const float tot = red[0] + red[1] + red[2] + red[3];
    const float inv = 1.0f / (sqrtf(tot) + 1e-8f);
    _Float16* dst = Xh + (size_t)row * D;
    for (int e = 0; e < 4; ++e) dst[tid + e * 128] = (_Float16)(v[e] * inv);
}

// ---------------------------------------------------------------------------
// K3: f32 -> f16 weight conversion
// ---------------------------------------------------------------------------
__global__ void f32_to_f16_kernel(const float* __restrict__ in,
                                  _Float16* __restrict__ out, int n)
{
    const int i = blockIdx.x * 256 + threadIdx.x;
    if (i < n) out[i] = (_Float16)in[i];
}

// ---------------------------------------------------------------------------
// K4: GEMM1  Hpre = X @ W1^T + b1   (R x H), f16 out, + deterministic
//     per-batch partial column sums / sumsq for BN statistics.
//     Workgroup tile: M=96 (one batch) x N=128; 8 waves, 16 cols each.
//     Double-buffered LDS, async global->LDS staging.
// ---------------------------------------------------------------------------
__global__ void __launch_bounds__(256)
gemm1_kernel(const _Float16* __restrict__ Xh, const _Float16* __restrict__ W1h,
             const float* __restrict__ bias1, _Float16* __restrict__ Hpre,
             float* __restrict__ sumP, float* __restrict__ sqP)
{
    constexpr int BK = 32, BN = 128, BM = 96;
    constexpr int NK = D / BK;                      // 16 k-chunks
    __shared__ __align__(16) _Float16 As[2][BM][BK];
    __shared__ __align__(16) _Float16 Bs[2][BN][BK];

    const int nslice = blockIdx.x;
    const int b      = blockIdx.y;
    const int tid    = threadIdx.x;
    const int wave   = tid >> 5;
    const int lane   = tid & 31;
    const int lan16  = lane & 15;
    const int hi     = lane >> 4;
    const int nbase  = nslice * BN;

    v8f acc[6];
    for (int t = 0; t < 6; ++t)
        for (int i = 0; i < 8; ++i) acc[t][i] = 0.0f;

    const _Float16* Xrow = Xh + (size_t)b * KTOK * D;

    auto stage = [&](int buf, int k) {
        for (int c = tid; c < BM * 4; c += 256) {
            const int r = c >> 2, ch = c & 3;
            async_cp16(&As[buf][r][ch * 8], &Xrow[(size_t)r * D + k + ch * 8]);
        }
        for (int c = tid; c < BN * 4; c += 256) {
            const int r = c >> 2, ch = c & 3;
            async_cp16(&Bs[buf][r][ch * 8], &W1h[(size_t)(nbase + r) * D + k + ch * 8]);
        }
    };

    stage(0, 0);
    stage_wait();

    for (int kt = 0; kt < NK; ++kt) {
        const int buf = kt & 1;
        if (kt + 1 < NK) stage(buf ^ 1, (kt + 1) * BK);

        // B fragment: lane -> B[K = j + (j>=8?8:0) + 8*hi][N = lane&15]
        v16h bfrag;
        {
            v8h bv0 = *(const v8h*)&Bs[buf][wave * 16 + lan16][hi * 8];
            v8h bv1 = *(const v8h*)&Bs[buf][wave * 16 + lan16][16 + hi * 8];
            for (int i = 0; i < 8; ++i) { bfrag[i] = bv0[i]; bfrag[i + 8] = bv1[i]; }
        }
        for (int mt = 0; mt < 6; ++mt) {
            // A fragment: lane -> A[M = lane&15][K = j + (j>=8?8:0) + 8*hi]
            v16h afrag;
            v8h a0 = *(const v8h*)&As[buf][mt * 16 + lan16][hi * 8];
            v8h a1 = *(const v8h*)&As[buf][mt * 16 + lan16][16 + hi * 8];
            for (int i = 0; i < 8; ++i) { afrag[i] = a0[i]; afrag[i + 8] = a1[i]; }
            acc[mt] = __builtin_amdgcn_wmma_f32_16x16x32_f16(
                false, afrag, false, bfrag, (short)0, acc[mt], false, false);
        }
        stage_wait();
    }

    // epilogue: add b1, store pre-BN H (f16), accumulate column partials
    const int n = nbase + wave * 16 + lan16;
    const float bia = bias1[n];
    float s1 = 0.f, s2 = 0.f;
    for (int mt = 0; mt < 6; ++mt) {
        for (int r = 0; r < 8; ++r) {
            const int m = mt * 16 + r + 8 * hi;
            const float h = acc[mt][r] + bia;
            s1 += h; s2 += h * h;
            Hpre[((size_t)(b * KTOK) + m) * H + n] = (_Float16)h;
        }
    }
    s1 += __shfl_xor(s1, 16, 32);
    s2 += __shfl_xor(s2, 16, 32);
    if (hi == 0) {
        sumP[(size_t)b * H + n] = s1;
        sqP[(size_t)b * H + n]  = s2;
    }
}

// ---------------------------------------------------------------------------
// K5: BN statistics -> per-column affine (a = gamma*invstd, b = beta - mu*a)
// ---------------------------------------------------------------------------
__global__ void bnstats_kernel(const float* __restrict__ sumP,
                               const float* __restrict__ sqP,
                               const float* __restrict__ gamma,
                               const float* __restrict__ beta,
                               float* __restrict__ Abn, float* __restrict__ Bbn)
{
    const int j = blockIdx.x * 256 + threadIdx.x;
    if (j >= H) return;
    float s1 = 0.f, s2 = 0.f;
    for (int b = 0; b < B; ++b) {
        s1 += sumP[(size_t)b * H + j];
        s2 += sqP[(size_t)b * H + j];
    }
    const float R   = (float)(B * KTOK);
    const float mu  = s1 / R;
    const float var = s2 / R - mu * mu;
    const float inv = rsqrtf(var + 1e-5f);
    const float a   = gamma[j] * inv;
    Abn[j] = a;
    Bbn[j] = beta[j] - mu * a;
}

// ---------------------------------------------------------------------------
// K5b: apply BN+ReLU to Hpre in place (f16), so GEMM2 staging is a pure copy
// ---------------------------------------------------------------------------
__global__ void __launch_bounds__(256)
bnrelu_kernel(_Float16* __restrict__ Hpre,
              const float* __restrict__ Abn, const float* __restrict__ Bbn)
{
    const size_t i = ((size_t)blockIdx.x * 256 + threadIdx.x) * 8;  // element idx
    const int kk = (int)(i % H);                                    // H % 8 == 0
    v8h v = *(v8h*)&Hpre[i];
    v8h o;
    for (int e = 0; e < 8; ++e) {
        const float h = (float)v[e] * Abn[kk + e] + Bbn[kk + e];
        o[e] = (_Float16)fmaxf(h, 0.0f);
    }
    *(v8h*)&Hpre[i] = o;
}

// ---------------------------------------------------------------------------
// K6: GEMM2  out = Hact @ W2^T, fused max-pool over the 96 tokens.
//     Double-buffered LDS, async global->LDS staging.
// ---------------------------------------------------------------------------
__global__ void __launch_bounds__(256)
gemm2_kernel(const _Float16* __restrict__ Hact, const _Float16* __restrict__ W2h,
             const float* __restrict__ bias2, float* __restrict__ pooled)
{
    constexpr int BK = 32, BN = 128, BM = 96;
    constexpr int NK = H / BK;                      // 32 k-chunks
    __shared__ __align__(16) _Float16 As[2][BM][BK];
    __shared__ __align__(16) _Float16 Bs[2][BN][BK];

    const int nslice = blockIdx.x;        // 0..15 (128-wide E slice)
    const int b      = blockIdx.y;
    const int tid    = threadIdx.x;
    const int wave   = tid >> 5;
    const int lane   = tid & 31;
    const int lan16  = lane & 15;
    const int hi     = lane >> 4;
    const int nbase  = nslice * BN;

    v8f acc[6];
    for (int t = 0; t < 6; ++t)
        for (int i = 0; i < 8; ++i) acc[t][i] = 0.0f;

    const _Float16* Hrow = Hact + (size_t)b * KTOK * H;

    auto stage = [&](int buf, int k) {
        for (int c = tid; c < BM * 4; c += 256) {
            const int r = c >> 2, ch = c & 3;
            async_cp16(&As[buf][r][ch * 8], &Hrow[(size_t)r * H + k + ch * 8]);
        }
        for (int c = tid; c < BN * 4; c += 256) {
            const int r = c >> 2, ch = c & 3;
            async_cp16(&Bs[buf][r][ch * 8], &W2h[(size_t)(nbase + r) * H + k + ch * 8]);
        }
    };

    stage(0, 0);
    stage_wait();

    for (int kt = 0; kt < NK; ++kt) {
        const int buf = kt & 1;
        if (kt + 1 < NK) stage(buf ^ 1, (kt + 1) * BK);

        v16h bfrag;
        {
            v8h bv0 = *(const v8h*)&Bs[buf][wave * 16 + lan16][hi * 8];
            v8h bv1 = *(const v8h*)&Bs[buf][wave * 16 + lan16][16 + hi * 8];
            for (int i = 0; i < 8; ++i) { bfrag[i] = bv0[i]; bfrag[i + 8] = bv1[i]; }
        }
        for (int mt = 0; mt < 6; ++mt) {
            v16h afrag;
            v8h a0 = *(const v8h*)&As[buf][mt * 16 + lan16][hi * 8];
            v8h a1 = *(const v8h*)&As[buf][mt * 16 + lan16][16 + hi * 8];
            for (int i = 0; i < 8; ++i) { afrag[i] = a0[i]; afrag[i + 8] = a1[i]; }
            acc[mt] = __builtin_amdgcn_wmma_f32_16x16x32_f16(
                false, afrag, false, bfrag, (short)0, acc[mt], false, false);
        }
        stage_wait();
    }

    // epilogue: max over the 96 token rows, then + b2
    const int n = nbase + wave * 16 + lan16;
    float mx = -FLT_MAX;
    for (int mt = 0; mt < 6; ++mt)
        for (int r = 0; r < 8; ++r) mx = fmaxf(mx, acc[mt][r]);
    mx = fmaxf(mx, __shfl_xor(mx, 16, 32));
    if (hi == 0) pooled[(size_t)b * E + n] = mx + bias2[n];
}

// ---------------------------------------------------------------------------
// K7: per-group dynamic projection + l2norm + final add (small, scalar FMA)
// ---------------------------------------------------------------------------
__global__ void __launch_bounds__(256)
dynproj_kernel(const float* __restrict__ bf, const int* __restrict__ idx,
               const float* __restrict__ Wdyn, const float* __restrict__ bdyn,
               const float* __restrict__ Wlin, const float* __restrict__ blin,
               const float* __restrict__ pooled, float* __restrict__ out)
{
    __shared__ float wdyn_s[G4 * D];          // 8 KB
    __shared__ float y[G4][RG];               // 6 KB
    __shared__ float nf[G4][E];               // 32 KB
    __shared__ float red[G4][8];
    __shared__ float norminv[G4];

    const int g   = blockIdx.x;               // 0..127
    const int tid = threadIdx.x;
    const int wave = tid >> 5, lane = tid & 31;

    for (int i = tid; i < G4 * D; i += 256) wdyn_s[i] = Wdyn[i];
    __syncthreads();

    // y[r][i] = Wdyn[r,:] . feats(group row i) + bdyn[r]
    for (int i = tid; i < RG; i += 256) {
        const int bb  = g * G4 + i / KTOK;
        const int tok = idx[bb * KTOK + (i % KTOK)];
        const float* row = bf + ((size_t)bb * N + tok) * D;
        float a0 = bdyn[0], a1 = bdyn[1], a2 = bdyn[2], a3 = bdyn[3];
        for (int d = 0; d < D; ++d) {
            const float x = row[d];
            a0 += x * wdyn_s[0 * D + d];
            a1 += x * wdyn_s[1 * D + d];
            a2 += x * wdyn_s[2 * D + d];
            a3 += x * wdyn_s[3 * D + d];
        }
        y[0][i] = a0; y[1][i] = a1; y[2][i] = a2; y[3][i] = a3;
    }
    __syncthreads();

    // nf[r][e] = y[r,:] . Wlin[e,:] + blin[e]
    float ss[G4] = {0.f, 0.f, 0.f, 0.f};
    for (int e = tid; e < E; e += 256) {
        const float* wl = Wlin + (size_t)e * RG;
        float a0 = blin[e], a1 = a0, a2 = a0, a3 = a0;
        for (int i = 0; i < RG; ++i) {
            const float w = wl[i];
            a0 += y[0][i] * w;
            a1 += y[1][i] * w;
            a2 += y[2][i] * w;
            a3 += y[3][i] * w;
        }
        nf[0][e] = a0; nf[1][e] = a1; nf[2][e] = a2; nf[3][e] = a3;
        ss[0] += a0 * a0; ss[1] += a1 * a1; ss[2] += a2 * a2; ss[3] += a3 * a3;
    }
    // block-wide sum of squares per output row
    for (int r = 0; r < G4; ++r) {
        float v = ss[r];
        for (int o = 16; o > 0; o >>= 1) v += __shfl_xor(v, o, 32);
        if (lane == 0) red[r][wave] = v;
    }
    __syncthreads();
    if (tid < G4) {
        float t = 0.f;
        for (int w = 0; w < 8; ++w) t += red[tid][w];
        norminv[tid] = 1.0f / (sqrtf(t) + 1e-8f);
    }
    __syncthreads();

    for (int e = tid; e < E; e += 256) {
        for (int r = 0; r < G4; ++r) {
            const int bb = g * G4 + r;
            out[(size_t)bb * E + e] = pooled[(size_t)bb * E + e] + nf[r][e] * norminv[r];
        }
    }
}

// ---------------------------------------------------------------------------
// Host launcher
// ---------------------------------------------------------------------------
static inline size_t align_up(size_t x, size_t a) { return (x + a - 1) & ~(a - 1); }

extern "C" void kernel_launch(void* const* d_in, const int* in_sizes, int n_in,
                              void* d_out, int out_size, void* d_ws, size_t ws_size,
                              hipStream_t stream)
{
    (void)in_sizes; (void)n_in; (void)out_size; (void)ws_size;

    const float* base_features = (const float*)d_in[0];
    const float* atten         = (const float*)d_in[1];
    // d_in[2] = pid (contiguous groups assumed, as in the reference reshape)
    const float* W_dyn  = (const float*)d_in[3];
    const float* b_dyn  = (const float*)d_in[4];
    const float* W_lin  = (const float*)d_in[5];
    const float* b_lin  = (const float*)d_in[6];
    const float* W1     = (const float*)d_in[7];
    const float* b1     = (const float*)d_in[8];
    const float* gamma1 = (const float*)d_in[9];
    const float* beta1  = (const float*)d_in[10];
    const float* W2     = (const float*)d_in[11];
    const float* b2     = (const float*)d_in[12];
    float* out = (float*)d_out;

    char* ws = (char*)d_ws;
    size_t off = 0;
    auto carve = [&](size_t bytes) {
        size_t o = align_up(off, 256);
        off = o + bytes;
        return (void*)(ws + o);
    };
    int*      idx    = (int*)     carve((size_t)B * KTOK * sizeof(int));
    _Float16* Xh     = (_Float16*)carve((size_t)B * KTOK * D * sizeof(_Float16));
    _Float16* W1h    = (_Float16*)carve((size_t)H * D * sizeof(_Float16));
    _Float16* W2h    = (_Float16*)carve((size_t)E * H * sizeof(_Float16));
    _Float16* Hpre   = (_Float16*)carve((size_t)B * KTOK * H * sizeof(_Float16));
    float*    sumP   = (float*)   carve((size_t)B * H * sizeof(float));
    float*    sqP    = (float*)   carve((size_t)B * H * sizeof(float));
    float*    Abn    = (float*)   carve((size_t)H * sizeof(float));
    float*    Bbn    = (float*)   carve((size_t)H * sizeof(float));
    float*    pooled = (float*)   carve((size_t)B * E * sizeof(float));

    // 1. top-k selection
    hipLaunchKernelGGL(topk_kernel, dim3(B), dim3(256), 0, stream, atten, idx);
    // 2. gather + l2norm -> f16 activations
    hipLaunchKernelGGL(gather_norm_kernel, dim3(B * KTOK), dim3(128), 0, stream,
                       base_features, idx, Xh);
    // 3. weight conversion
    hipLaunchKernelGGL(f32_to_f16_kernel, dim3((H * D + 255) / 256), dim3(256), 0, stream,
                       W1, W1h, H * D);
    hipLaunchKernelGGL(f32_to_f16_kernel, dim3((E * H + 255) / 256), dim3(256), 0, stream,
                       W2, W2h, E * H);
    // 4. GEMM1 (WMMA, async double-buffered) + BN partial stats
    hipLaunchKernelGGL(gemm1_kernel, dim3(H / 128, B), dim3(256), 0, stream,
                       Xh, W1h, b1, Hpre, sumP, sqP);
    // 5. BN statistics
    hipLaunchKernelGGL(bnstats_kernel, dim3((H + 255) / 256), dim3(256), 0, stream,
                       sumP, sqP, gamma1, beta1, Abn, Bbn);
    // 5b. apply BN+ReLU in place
    hipLaunchKernelGGL(bnrelu_kernel,
                       dim3((int)(((size_t)B * KTOK * H / 8) / 256)), dim3(256), 0, stream,
                       Hpre, Abn, Bbn);
    // 6. GEMM2 (WMMA, async double-buffered) with fused token max-pool
    hipLaunchKernelGGL(gemm2_kernel, dim3(E / 128, B), dim3(256), 0, stream,
                       Hpre, W2h, b2, pooled);
    // 7. dynamic projection + l2norm + final add
    hipLaunchKernelGGL(dynproj_kernel, dim3(GRP), dim3(256), 0, stream,
                       base_features, idx, W_dyn, b_dyn, W_lin, b_lin, pooled, out);
}
